// CausalSelfAttentionHead_27745488732245
// MI455X (gfx1250) — matrix operand order repaired and checked by reading, not compiled, not run on hardware
//
#include <hip/hip_runtime.h>
#include <hip/hip_bf16.h>

// Problem constants (from the reference)
constexpr int NBATCH = 4;
constexpr int T      = 2048;
constexpr int DM     = 768;   // d_model
constexpr int H      = 64;    // head_dim
constexpr int ROWS   = NBATCH * T;           // 8192 flat rows
constexpr int XS_STRIDE_H = DM + 8;          // 776 halves -> conflict-free A reads
constexpr int P_STRIDE_H  = 40;              // padded P-tile stride (halves)

typedef __attribute__((ext_vector_type(16))) __bf16 v16bf;
typedef __attribute__((ext_vector_type(8)))  float  v8f;

union Frag {
    v16bf    v;
    unsigned u[8];
};

__device__ inline unsigned short f32_to_bf16(float f) {
    unsigned u = __float_as_uint(f);
    unsigned r = u + 0x7FFFu + ((u >> 16) & 1u);   // round-to-nearest-even
    return (unsigned short)(r >> 16);
}

__device__ inline unsigned pack2_bf16(float a, float b) {
    return (unsigned)f32_to_bf16(a) | ((unsigned)f32_to_bf16(b) << 16);
}

__device__ inline void fill_frag(Frag& f, uint4 a, uint4 b) {
    f.u[0] = a.x; f.u[1] = a.y; f.u[2] = a.z; f.u[3] = a.w;
    f.u[4] = b.x; f.u[5] = b.y; f.u[6] = b.z; f.u[7] = b.w;
}

__device__ inline v8f wmma_bf16(v16bf a, v16bf b, v8f c) {
    // D = A(16x32 bf16) * B(32x16 bf16) + C(16x16 f32)
    return __builtin_amdgcn_wmma_f32_16x16x32_bf16(
        /*neg_a=*/false, a, /*neg_b=*/false, b,
        /*c_mod=*/(short)0, c, /*reuse_a=*/false, /*reuse_b=*/false);
}

// ---------------------------------------------------------------------------
// Kernel 0: W[768][64] fp32 -> W^T[64][768] bf16 (B-operand-natural layout)
// ---------------------------------------------------------------------------
__global__ __launch_bounds__(256)
void wt_convert_kernel(const float* __restrict__ Wq,
                       const float* __restrict__ Wk,
                       const float* __restrict__ Wv,
                       unsigned short* __restrict__ WTq,
                       unsigned short* __restrict__ WTk,
                       unsigned short* __restrict__ WTv)
{
    int i = blockIdx.x * 256 + threadIdx.x;      // over 64*768 elements
    if (i >= H * DM) return;
    int n = i / DM, d = i % DM;
    WTq[i] = f32_to_bf16(Wq[(size_t)d * H + n]);
    WTk[i] = f32_to_bf16(Wk[(size_t)d * H + n]);
    WTv[i] = f32_to_bf16(Wv[(size_t)d * H + n]);
}

// ---------------------------------------------------------------------------
// Kernel 1: q/k/v = x @ W{q,k,v} + b  -> bf16, v stored transposed (h-major)
// One WG (4 waves) per 16-row tile; wave w owns output columns [16w,16w+16).
// ---------------------------------------------------------------------------
__global__ __launch_bounds__(128)
void qkv_proj_kernel(const float* __restrict__ x,
                     const float* __restrict__ bq,
                     const float* __restrict__ bk,
                     const float* __restrict__ bv,
                     const unsigned short* __restrict__ WTq,
                     const unsigned short* __restrict__ WTk,
                     const unsigned short* __restrict__ WTv,
                     unsigned short* __restrict__ qB,
                     unsigned short* __restrict__ kB,
                     unsigned short* __restrict__ vT)
{
    __shared__ __align__(16) unsigned xs[(16 * XS_STRIDE_H) / 2]; // 16x768 bf16

    const int tid    = threadIdx.x;
    const int warp   = tid >> 5;
    const int lane   = tid & 31;
    const int lane16 = lane & 15;
    const int hs     = lane >> 4;          // lane half: 0 or 1
    const int row0   = blockIdx.x * 16;    // tiles never straddle batches

    // Stage x tile into LDS as bf16 (float4 reads, packed uint2 stores)
    for (int i = tid; i < 16 * (DM / 4); i += 128) {
        int r  = i / (DM / 4);
        int c4 = (i % (DM / 4)) * 4;
        float4 xv = *(const float4*)(x + (size_t)(row0 + r) * DM + c4);
        uint2 pk;
        pk.x = pack2_bf16(xv.x, xv.y);
        pk.y = pack2_bf16(xv.z, xv.w);
        *(uint2*)&xs[r * (XS_STRIDE_H / 2) + (c4 >> 1)] = pk;
    }
    __syncthreads();

    const int n = warp * 16 + lane16;      // this wave's output column
    const unsigned* xsrow = xs + lane16 * (XS_STRIDE_H / 2);
    const unsigned* wq = (const unsigned*)WTq + (size_t)n * (DM / 2);
    const unsigned* wk = (const unsigned*)WTk + (size_t)n * (DM / 2);
    const unsigned* wv = (const unsigned*)WTv + (size_t)n * (DM / 2);

    v8f accq = {}, acck = {}, accv = {};

    for (int kc = 0; kc < DM / 32; ++kc) {
        Frag A, Bq, Bk, Bv;
        // A-frag: dwords kc*16 + 8*hs + j  (8 consecutive)
        fill_frag(A, *(const uint4*)(xsrow + kc * 16 + 8 * hs),
                     *(const uint4*)(xsrow + kc * 16 + 8 * hs + 4));
        // B-frag: dwords kc*16 + 8*hs + j  (8 consecutive)
        fill_frag(Bq, *(const uint4*)(wq + kc * 16 + 8 * hs),
                      *(const uint4*)(wq + kc * 16 + 8 * hs + 4));
        fill_frag(Bk, *(const uint4*)(wk + kc * 16 + 8 * hs),
                      *(const uint4*)(wk + kc * 16 + 8 * hs + 4));
        fill_frag(Bv, *(const uint4*)(wv + kc * 16 + 8 * hs),
                      *(const uint4*)(wv + kc * 16 + 8 * hs + 4));
        accq = wmma_bf16(A.v, Bq.v, accq);
        acck = wmma_bf16(A.v, Bk.v, acck);
        accv = wmma_bf16(A.v, Bv.v, accv);
    }

    const float biasq = bq[n], biask = bk[n], biasv = bv[n];
    #pragma unroll
    for (int g = 0; g < 8; ++g) {
        int grow = row0 + g + 8 * hs;              // C/D layout: row = g + 8*half
        qB[(size_t)grow * H + n] = f32_to_bf16(accq[g] + biasq);
        kB[(size_t)grow * H + n] = f32_to_bf16(acck[g] + biask);
        int b = grow >> 11, t = grow & (T - 1);
        vT[((size_t)b * H + n) * T + t] = f32_to_bf16(accv[g] + biasv);
    }
}

// ---------------------------------------------------------------------------
// Kernel 2: flash attention. One wave per 16-query tile; waves independent.
// Per 32-key tile: S=Q.K^T (4 wmma) -> masked softmax -> P.V (4 wmma)
// + 1 wmma against an all-ones B fragment to get row-sums (no shuffle sums).
// Only the single diagonal tile applies the causal mask.
// ---------------------------------------------------------------------------
__global__ __launch_bounds__(128)
void flash_attn_kernel(const unsigned short* __restrict__ qB,
                       const unsigned short* __restrict__ kB,
                       const unsigned short* __restrict__ vT,
                       float* __restrict__ out)
{
    __shared__ __align__(16) unsigned short plds[4][16 * P_STRIDE_H];

    const int tid    = threadIdx.x;
    const int warp   = tid >> 5;
    const int lane   = tid & 31;
    const int lane16 = lane & 15;
    const int hs     = lane >> 4;

    const int tile = blockIdx.x * 4 + warp;   // 512 query tiles total
    const int b    = tile >> 7;               // T/16 = 128 tiles per batch
    const int q0   = (tile & 127) << 4;

    const unsigned* qp = (const unsigned*)qB + (size_t)(b * T + q0) * (H / 2);
    const unsigned* kp = (const unsigned*)kB + (size_t)b * T * (H / 2);
    const unsigned* vp = (const unsigned*)vT + (size_t)b * H * (T / 2);

    // Q A-fragments for both 32-wide K chunks of H=64 (register resident)
    Frag aq[2];
    const unsigned* qrow = qp + lane16 * (H / 2);
    #pragma unroll
    for (int c = 0; c < 2; ++c)
        fill_frag(aq[c], *(const uint4*)(qrow + c * 16 + 8 * hs),
                         *(const uint4*)(qrow + c * 16 + 8 * hs + 4));

    // All-ones B fragment (bf16 1.0 pairs) for row-sum WMMA
    Frag onesB;
    #pragma unroll
    for (int j = 0; j < 8; ++j) onesB.u[j] = 0x3F803F80u;

    float mrow[8];
    v8f o[4] = {{}, {}, {}, {}};
    v8f osum = {};
    #pragma unroll
    for (int g = 0; g < 8; ++g) mrow[g] = -3.0e38f;

    const int nkeys = q0 + 16;                 // causal bound for this tile
    const int jb    = ((nkeys - 1) >> 5) << 5; // the single diagonal (masked) tile

    auto tile_body = [&](int j0, bool masked) {
        const unsigned* krow0 = kp + (size_t)(j0 + lane16)      * (H / 2);
        const unsigned* krow1 = kp + (size_t)(j0 + 16 + lane16) * (H / 2);

        // Prefetch next key tile (K rows + V^T line) into caches
        if (j0 + 32 < nkeys) {
            __builtin_prefetch(krow0 + 32 * (H / 2), 0, 1);
            __builtin_prefetch(krow1 + 32 * (H / 2), 0, 1);
            __builtin_prefetch(vp + (size_t)lane16 * (T / 2) + ((j0 + 32) >> 1), 0, 1);
        }

        // ---- S = Q @ K^T for 32 keys (two 16-key N halves) ----
        v8f s0 = {}, s1 = {};
        #pragma unroll
        for (int c = 0; c < 2; ++c) {
            Frag B0, B1;
            fill_frag(B0, *(const uint4*)(krow0 + c * 16 + 8 * hs),
                          *(const uint4*)(krow0 + c * 16 + 8 * hs + 4));
            fill_frag(B1, *(const uint4*)(krow1 + c * 16 + 8 * hs),
                          *(const uint4*)(krow1 + c * 16 + 8 * hs + 4));
            s0 = wmma_bf16(aq[c].v, B0.v, s0);
            s1 = wmma_bf16(aq[c].v, B1.v, s1);
        }

        // ---- scale (+ causal mask on diagonal tile) + online max ----
        float p0[8], p1[8], cf[8];
        #pragma unroll
        for (int g = 0; g < 8; ++g) {
            float v0, v1;
            if (masked) {
                int grow = q0 + g + 8 * hs;
                v0 = (j0 + lane16      <= grow) ? s0[g] * 0.125f : -3.0e38f;
                v1 = (j0 + 16 + lane16 <= grow) ? s1[g] * 0.125f : -3.0e38f;
            } else {
                v0 = s0[g] * 0.125f;
                v1 = s1[g] * 0.125f;
            }
            float tmax = fmaxf(v0, v1);
            #pragma unroll
            for (int m = 1; m < 16; m <<= 1)
                tmax = fmaxf(tmax, __shfl_xor(tmax, m, 32));   // within 16-lane half
            float mnew = fmaxf(mrow[g], tmax);
            cf[g] = __expf(mrow[g] - mnew);
            mrow[g] = mnew;
            p0[g] = __expf(v0 - mnew);
            p1[g] = __expf(v1 - mnew);
        }
        // rescale output + row-sum accumulators
        #pragma unroll
        for (int s = 0; s < 4; ++s)
            #pragma unroll
            for (int g = 0; g < 8; ++g)
                o[s][g] *= cf[g];
        #pragma unroll
        for (int g = 0; g < 8; ++g)
            osum[g] *= cf[g];

        // ---- C-layout P -> LDS -> A-layout fragment (per-wave region) ----
        #pragma unroll
        for (int g = 0; g < 8; ++g) {
            int rl = g + 8 * hs;
            plds[warp][rl * P_STRIDE_H + lane16]      = f32_to_bf16(p0[g]);
            plds[warp][rl * P_STRIDE_H + 16 + lane16] = f32_to_bf16(p1[g]);
        }
        Frag ap;
        const unsigned* pp = (const unsigned*)&plds[warp][0] + lane16 * (P_STRIDE_H / 2);
        fill_frag(ap, *(const uint4*)(pp + 8 * hs),
                      *(const uint4*)(pp + 8 * hs + 4));

        // ---- O += P @ V ; row-sums += P @ ones ----
        #pragma unroll
        for (int s = 0; s < 4; ++s) {
            Frag Bv;
            const unsigned* vrow = vp + (size_t)(s * 16 + lane16) * (T / 2) + (j0 >> 1);
            fill_frag(Bv, *(const uint4*)(vrow + 8 * hs),
                          *(const uint4*)(vrow + 8 * hs + 4));
            o[s] = wmma_bf16(ap.v, Bv.v, o[s]);
        }
        osum = wmma_bf16(ap.v, onesB.v, osum);
    };

    for (int j0 = 0; j0 < jb; j0 += 32) tile_body(j0, false);  // interior tiles
    tile_body(jb, true);                                       // diagonal tile

    // ---- epilogue: divide by row sum, write fp32 ----
    float inv[8];
    #pragma unroll
    for (int g = 0; g < 8; ++g) inv[g] = 1.0f / osum[g];
    #pragma unroll
    for (int s = 0; s < 4; ++s)
        #pragma unroll
        for (int g = 0; g < 8; ++g) {
            int grow = q0 + g + 8 * hs;
            out[(size_t)(b * T + grow) * H + s * 16 + lane16] = o[s][g] * inv[g];
        }
}

// ---------------------------------------------------------------------------
extern "C" void kernel_launch(void* const* d_in, const int* in_sizes, int n_in,
                              void* d_out, int out_size, void* d_ws, size_t ws_size,
                              hipStream_t stream)
{
    const float* x  = (const float*)d_in[0];
    const float* Wq = (const float*)d_in[1];
    const float* bq = (const float*)d_in[2];
    const float* Wk = (const float*)d_in[3];
    const float* bk = (const float*)d_in[4];
    const float* Wv = (const float*)d_in[5];
    const float* bv = (const float*)d_in[6];

    unsigned short* qB  = (unsigned short*)d_ws;           // [8192][64] bf16
    unsigned short* kB  = qB  + (size_t)ROWS * H;          // [8192][64] bf16
    unsigned short* vT  = kB  + (size_t)ROWS * H;          // [4][64][2048] bf16
    unsigned short* WTq = vT  + (size_t)NBATCH * H * T;    // [64][768] bf16
    unsigned short* WTk = WTq + (size_t)H * DM;
    unsigned short* WTv = WTk + (size_t)H * DM;

    wt_convert_kernel<<<(H * DM + 255) / 256, 256, 0, stream>>>(Wq, Wk, Wv,
                                                                WTq, WTk, WTv);
    qkv_proj_kernel<<<ROWS / 16, 128, 0, stream>>>(x, bq, bk, bv,
                                                   WTq, WTk, WTv, qB, kB, vT);
    flash_attn_kernel<<<ROWS / 64, 128, 0, stream>>>(qB, kB, vT, (float*)d_out);
}
